// Model_2508260901903
// MI455X (gfx1250) — matrix operand order repaired and checked by reading, not compiled, not run on hardware
//
#include <hip/hip_runtime.h>
#include <cstdint>
#include <cstddef>

// ---------------------------------------------------------------------------
// Longformer-style encoder forward for MI455X (gfx1250), bf16 WMMA everywhere.
// S=4096 D=768 H=12 DH=64 F=3072 L=12 W=256 (one-sided), B=1, C=10.
// ---------------------------------------------------------------------------

#define S_LEN 4096
#define D_DIM 768
#define F_DIM 3072
#define C_CLS 10

typedef __attribute__((ext_vector_type(16))) __bf16 v16bf;
typedef __attribute__((ext_vector_type(8)))  float  v8f;
typedef __attribute__((ext_vector_type(4)))  int    v4i;

union Frag { unsigned int u[8]; v16bf v; };

#if defined(__has_builtin)
#if __has_builtin(__builtin_amdgcn_global_load_async_to_lds_b128)
#define HAVE_ASYNC_LDS 1
#endif
#endif

__device__ __forceinline__ unsigned int pack_bf16(float a, float b) {
  unsigned ua = __float_as_uint(a); ua += 0x7fffu + ((ua >> 16) & 1u);
  unsigned ub = __float_as_uint(b); ub += 0x7fffu + ((ub >> 16) & 1u);
  return (ua >> 16) | (ub & 0xffff0000u);
}
__device__ __forceinline__ unsigned short f2bf(float a) {
  unsigned ua = __float_as_uint(a); ua += 0x7fffu + ((ua >> 16) & 1u);
  return (unsigned short)(ua >> 16);
}
__device__ __forceinline__ v8f wmma_bf16(v16bf a, v16bf b, v8f c) {
  return __builtin_amdgcn_wmma_f32_16x16x32_bf16(false, a, false, b, (short)0, c,
                                                 false, false);
}

// Copy 16 bytes global(bf16) -> LDS. Uses CDNA5 async-to-LDS when available.
__device__ __forceinline__ void copy16(unsigned int* l, const unsigned short* g) {
#if HAVE_ASYNC_LDS
  __builtin_amdgcn_global_load_async_to_lds_b128(
      (__attribute__((address_space(1))) v4i*)g,
      (__attribute__((address_space(3))) v4i*)l, 0, 0);
#else
  uint4 t = *(const uint4*)g;
  l[0] = t.x; l[1] = t.y; l[2] = t.z; l[3] = t.w;
#endif
}
__device__ __forceinline__ void async_wait0() {
#if HAVE_ASYNC_LDS
#if __has_builtin(__builtin_amdgcn_s_wait_asynccnt)
  __builtin_amdgcn_s_wait_asynccnt(0);
#else
  asm volatile("s_wait_asynccnt 0x0" ::: "memory");
#endif
#endif
}

// A-fragment dword (K-pair) index (16-bit A 16x32), same k order as B:
//   lanes 0-15: V0..7 -> K0..15 ; lanes 16-31: K16..31
#define APAIR(j, lh) ((j) + (lh) * 8)
// B-fragment dword index (16-bit B 32x16): lanes 0-15 K0..15, lanes 16-31 K16..31
#define BPAIR(j, lh) ((j) + (lh) * 8)

// ---------------------------------------------------------------------------
__device__ __forceinline__ float block_reduce_sum(float v, float* sbuf) {
#pragma unroll
  for (int off = 16; off > 0; off >>= 1) v += __shfl_xor(v, off, 32);
  __syncthreads();
  if ((threadIdx.x & 31) == 0) sbuf[threadIdx.x >> 5] = v;
  __syncthreads();
  float tot = 0.f;
#pragma unroll
  for (int i = 0; i < 8; i++) tot += sbuf[i];
  return tot;
}

// ---------------------------------------------------------------------------
// Embedding gather + LayerNorm. One block per token. Writes f32 + bf16.
// ---------------------------------------------------------------------------
__global__ __launch_bounds__(256) void embed_ln_kernel(
    const int* __restrict__ ctx, const float* __restrict__ wemb,
    const float* __restrict__ pemb, const float* __restrict__ g,
    const float* __restrict__ bb, float* __restrict__ xf,
    unsigned short* __restrict__ xb) {
  __shared__ float sbuf[8];
  const int s = blockIdx.x;
  const int tok = ctx[s];
  float v[3];
  float sum = 0.f;
#pragma unroll
  for (int i = 0; i < 3; i++) {
    int d = threadIdx.x + i * 256;
    v[i] = wemb[(size_t)tok * D_DIM + d] + pemb[(size_t)s * D_DIM + d];
    sum += v[i];
  }
  float mean = block_reduce_sum(sum, sbuf) * (1.f / 768.f);
  float var = 0.f;
#pragma unroll
  for (int i = 0; i < 3; i++) { float t = v[i] - mean; var += t * t; }
  var = block_reduce_sum(var, sbuf) * (1.f / 768.f);
  float rstd = rsqrtf(var + 1e-5f);
#pragma unroll
  for (int i = 0; i < 3; i++) {
    int d = threadIdx.x + i * 256;
    float o = (v[i] - mean) * rstd * g[d] + bb[d];
    xf[(size_t)s * D_DIM + d] = o;
    xb[(size_t)s * D_DIM + d] = f2bf(o);
  }
}

// ---------------------------------------------------------------------------
// Residual add + LayerNorm. One block per token. x = LN(x + y).
// ---------------------------------------------------------------------------
__global__ __launch_bounds__(256) void add_ln_kernel(
    float* __restrict__ xf, const float* __restrict__ yf,
    const float* __restrict__ g, const float* __restrict__ bb,
    unsigned short* __restrict__ xb) {
  __shared__ float sbuf[8];
  const int s = blockIdx.x;
  float v[3];
  float sum = 0.f;
#pragma unroll
  for (int i = 0; i < 3; i++) {
    int d = threadIdx.x + i * 256;
    v[i] = xf[(size_t)s * D_DIM + d] + yf[(size_t)s * D_DIM + d];
    sum += v[i];
  }
  float mean = block_reduce_sum(sum, sbuf) * (1.f / 768.f);
  float var = 0.f;
#pragma unroll
  for (int i = 0; i < 3; i++) { float t = v[i] - mean; var += t * t; }
  var = block_reduce_sum(var, sbuf) * (1.f / 768.f);
  float rstd = rsqrtf(var + 1e-5f);
#pragma unroll
  for (int i = 0; i < 3; i++) {
    int d = threadIdx.x + i * 256;
    float o = (v[i] - mean) * rstd * g[d] + bb[d];
    xf[(size_t)s * D_DIM + d] = o;
    xb[(size_t)s * D_DIM + d] = f2bf(o);
  }
}

// ---------------------------------------------------------------------------
// Weight transpose + f32->bf16 convert: Wt[n*K + k] = bf16(W[k*N + n]).
// 32x32 tiles, 256 threads.
// ---------------------------------------------------------------------------
__global__ __launch_bounds__(256) void wconv_kernel(
    const float* __restrict__ W, unsigned short* __restrict__ Wt, int K, int N) {
  __shared__ float t[32][33];
  const int kb = blockIdx.x * 32, nb = blockIdx.y * 32;
  const int tx = threadIdx.x & 31, ty = threadIdx.x >> 5;  // ty 0..7
#pragma unroll
  for (int i = 0; i < 4; i++)
    t[ty + i * 8][tx] = W[(size_t)(kb + ty + i * 8) * N + nb + tx];
  __syncthreads();
#pragma unroll
  for (int i = 0; i < 4; i++) {
    int n = nb + ty + i * 8;
    Wt[(size_t)n * K + kb + tx] = f2bf(t[tx][ty + i * 8]);
  }
}

// ---------------------------------------------------------------------------
// Tiled bf16 WMMA GEMM: Y[M,N] = act(Xbf16[M,K] @ W + bias[N]) with weights
// pre-transposed to bf16 N-major (Wt[n*K+k]). Block tile 128x128, K-stage 64
// (two WMMA k-steps per stage). 8 waves in 2(M)x4(N); each wave owns
// 64x32 = 4x2 WMMA tiles. Double-buffered LDS stages filled with CDNA5
// async-load-to-LDS: stage i+1's copies are issued right after the barrier
// and land while stage i computes (one barrier per stage).
// ACT: 0=none, 1=exact GELU. OUTB -> bf16 output, OUTF -> f32 output.
// ---------------------------------------------------------------------------
template <int ACT, bool OUTB, bool OUTF>
__global__ __launch_bounds__(256) void gemm_kernel(
    const unsigned short* __restrict__ X, const unsigned short* __restrict__ Wt,
    const float* __restrict__ bias, float* __restrict__ outF,
    unsigned short* __restrict__ outB, int M, int N, int K) {
  __shared__ unsigned int As[2][128 * 33];  // [buf][m][k] 64 bf16 + pad
  __shared__ unsigned int Bs[2][128 * 33];  // [buf][n][k]
  const int tid = threadIdx.x;
  const int wave = tid >> 5, lane = tid & 31;
  const int lh = lane >> 4, li = lane & 15;
  const int wm = wave >> 2, wn = wave & 3;
  const int n0 = blockIdx.x * 128, m0 = blockIdx.y * 128;
  const int rowL = tid >> 1, sg = tid & 1;  // each thread stages 64B of A and B

  const unsigned short* gaBase = X + (size_t)(m0 + rowL) * K + sg * 32;
  const unsigned short* gbBase = Wt + (size_t)(n0 + rowL) * K + sg * 32;

  v8f acc[4][2];
#pragma unroll
  for (int i = 0; i < 4; i++)
#pragma unroll
    for (int j = 0; j < 2; j++)
#pragma unroll
      for (int r = 0; r < 8; r++) acc[i][j][r] = 0.f;

  // Stage 0 copies
  {
    unsigned int* la = &As[0][rowL * 33 + sg * 16];
    unsigned int* lb = &Bs[0][rowL * 33 + sg * 16];
#pragma unroll
    for (int c = 0; c < 4; c++) copy16(la + c * 4, gaBase + c * 8);
#pragma unroll
    for (int c = 0; c < 4; c++) copy16(lb + c * 4, gbBase + c * 8);
  }

  int ib = 0;
  for (int kt = 0; kt < K; kt += 64, ib ^= 1) {
    async_wait0();      // stage ib data has landed
    __syncthreads();    // visible to all waves; prior readers of buf ib^1 done
    if (kt + 64 < K) {  // overlap next stage's copies with this stage's math
      unsigned int* la = &As[ib ^ 1][rowL * 33 + sg * 16];
      unsigned int* lb = &Bs[ib ^ 1][rowL * 33 + sg * 16];
#pragma unroll
      for (int c = 0; c < 4; c++) copy16(la + c * 4, gaBase + kt + 64 + c * 8);
#pragma unroll
      for (int c = 0; c < 4; c++) copy16(lb + c * 4, gbBase + kt + 64 + c * 8);
      if (kt + 128 < K)  // lowers to global_prefetch_b8
        __builtin_prefetch(gbBase + kt + 128, 0, 1);
    }

#pragma unroll
    for (int ks = 0; ks < 2; ks++) {
      Frag a[4], b[2];
#pragma unroll
      for (int mt = 0; mt < 4; mt++) {
        const unsigned int* p = &As[ib][(wm * 64 + mt * 16 + li) * 33 + ks * 16];
#pragma unroll
        for (int j = 0; j < 8; j++) a[mt].u[j] = p[APAIR(j, lh)];
      }
#pragma unroll
      for (int nt = 0; nt < 2; nt++) {
        const unsigned int* p = &Bs[ib][(wn * 32 + nt * 16 + li) * 33 + ks * 16];
#pragma unroll
        for (int j = 0; j < 8; j++) b[nt].u[j] = p[BPAIR(j, lh)];
      }
#pragma unroll
      for (int mt = 0; mt < 4; mt++)
#pragma unroll
        for (int nt = 0; nt < 2; nt++)
          acc[mt][nt] = wmma_bf16(a[mt].v, b[nt].v, acc[mt][nt]);
    }
  }

  // Epilogue: C/D layout => VGPR r: lanes 0-15 M=r, lanes 16-31 M=r+8; N=li.
#pragma unroll
  for (int mt = 0; mt < 4; mt++) {
#pragma unroll
    for (int nt = 0; nt < 2; nt++) {
      int gn = n0 + wn * 32 + nt * 16 + li;
      float bvv = bias[gn];
#pragma unroll
      for (int r = 0; r < 8; r++) {
        int gm = m0 + wm * 64 + mt * 16 + r + lh * 8;
        float v = acc[mt][nt][r] + bvv;
        if (ACT == 1) v = 0.5f * v * (1.0f + erff(v * 0.70710678118654752f));
        if (OUTF) outF[(size_t)gm * N + gn] = v;
        if (OUTB) outB[(size_t)gm * N + gn] = f2bf(v);
      }
    }
  }
}

// ---------------------------------------------------------------------------
// Sliding-window flash attention. Block = 128 queries x 1 head (8 waves; each
// wave owns 16 query rows). Iterates 12 key tiles of 64 covering the 3W=768
// neighbor keys of the query chunk. Band/in-range/mask bias computed inline.
// ---------------------------------------------------------------------------
__global__ __launch_bounds__(256) void attn_kernel(
    const unsigned short* __restrict__ Qg, const unsigned short* __restrict__ Kg,
    const unsigned short* __restrict__ Vg, const int* __restrict__ mask,
    unsigned short* __restrict__ Og) {
  __shared__ unsigned int Qs[128 * 33];    // 128 x 64 bf16
  __shared__ unsigned int Ks[64 * 33];     // 64 keys x 64 d (row-major)
  __shared__ unsigned int Vs[64 * 33];     // V^T: 64 d x 64 keys
  __shared__ unsigned int Ps[8 * 16 * 33]; // per-wave 16 x 64 bf16 P tile
  const int tid = threadIdx.x;
  const int wave = tid >> 5, lane = tid & 31, lh = lane >> 4, li = lane & 15;
  const int q0 = blockIdx.x * 128;
  const int h = blockIdx.y;
  const int chunk = q0 >> 8;
  const int kbase = (chunk - 1) * 256;

  {  // Q tile (32 bf16 / thread)
    int r = tid >> 1, sg = tid & 1;
    const unsigned short* src = Qg + (size_t)(q0 + r) * D_DIM + h * 64 + sg * 32;
    unsigned int* dst = &Qs[r * 33 + sg * 16];
#pragma unroll
    for (int c = 0; c < 4; c++) copy16(dst + c * 4, src + c * 8);
  }

  v8f Oacc[4];
#pragma unroll
  for (int dt = 0; dt < 4; dt++)
#pragma unroll
    for (int r = 0; r < 8; r++) Oacc[dt][r] = 0.f;
  float mrow[8], lrow[8];
#pragma unroll
  for (int r = 0; r < 8; r++) { mrow[r] = -1e30f; lrow[r] = 0.f; }

  for (int kt = 0; kt < 12; kt++) {
    const int kg0 = kbase + kt * 64;
    __syncthreads();  // previous iteration's K/V consumers done
    {  // K tile: 64 keys x 64 d, zero-filled out of range
      int r = tid >> 2, sg4 = tid & 3;
      unsigned int* dst = &Ks[r * 33 + sg4 * 8];
      int kg = kg0 + r;
      if ((unsigned)kg < (unsigned)S_LEN) {
        const unsigned short* src = Kg + (size_t)kg * D_DIM + h * 64 + sg4 * 16;
        copy16(dst, src);
        copy16(dst + 4, src + 8);
      } else {
#pragma unroll
        for (int j = 0; j < 8; j++) dst[j] = 0u;
      }
    }
    {  // V tile transposed: Vs[d][k]
      int d = tid & 63, kp = tid >> 6;  // kp 0..3
      unsigned int* dst = &Vs[d * 33 + kp * 8];
#pragma unroll
      for (int j = 0; j < 8; j++) {
        int ka = kg0 + kp * 16 + 2 * j;
        unsigned int v0 = ((unsigned)ka < (unsigned)S_LEN)
                              ? (unsigned)Vg[(size_t)ka * D_DIM + h * 64 + d] : 0u;
        unsigned int v1 = ((unsigned)(ka + 1) < (unsigned)S_LEN)
                              ? (unsigned)Vg[(size_t)(ka + 1) * D_DIM + h * 64 + d] : 0u;
        dst[j] = v0 | (v1 << 16);
      }
    }
    async_wait0();
    __syncthreads();

    // S = Q @ K^T : M=16 (wave rows), N=64 keys, K=64 d (2 WMMA k-steps)
    v8f sacc[4];
#pragma unroll
    for (int nt = 0; nt < 4; nt++)
#pragma unroll
      for (int r = 0; r < 8; r++) sacc[nt][r] = 0.f;
#pragma unroll
    for (int ks = 0; ks < 2; ks++) {
      Frag a;
      const unsigned int* pa = &Qs[(wave * 16 + li) * 33 + ks * 16];
#pragma unroll
      for (int j = 0; j < 8; j++) a.u[j] = pa[APAIR(j, lh)];
#pragma unroll
      for (int nt = 0; nt < 4; nt++) {
        Frag b;
        const unsigned int* pb = &Ks[(nt * 16 + li) * 33 + ks * 16];
#pragma unroll
        for (int j = 0; j < 8; j++) b.u[j] = pb[BPAIR(j, lh)];
        sacc[nt] = wmma_bf16(a.v, b.v, sacc[nt]);
      }
    }

    // Bias (band / in-range / padding mask) + online softmax
    float rmax[8];
#pragma unroll
    for (int r = 0; r < 8; r++) rmax[r] = -1e30f;
#pragma unroll
    for (int nt = 0; nt < 4; nt++) {
      int kg = kg0 + nt * 16 + li;
      bool kok = ((unsigned)kg < (unsigned)S_LEN) && (mask[kg] != 0);
#pragma unroll
      for (int r = 0; r < 8; r++) {
        int qg = q0 + wave * 16 + r + lh * 8;
        int dlt = kg - qg;
        bool ok = kok && (dlt <= 256) && (dlt >= -256);
        float sv = sacc[nt][r] * 0.125f + (ok ? 0.f : -1e9f);
        sacc[nt][r] = sv;
        rmax[r] = fmaxf(rmax[r], sv);
      }
    }
#pragma unroll
    for (int o = 1; o < 16; o <<= 1)
#pragma unroll
      for (int r = 0; r < 8; r++)
        rmax[r] = fmaxf(rmax[r], __shfl_xor(rmax[r], o, 32));
    float fac[8], rsum[8];
#pragma unroll
    for (int r = 0; r < 8; r++) {
      float mn = fmaxf(mrow[r], rmax[r]);
      fac[r] = __expf(mrow[r] - mn);
      mrow[r] = mn;
      rsum[r] = 0.f;
    }
#pragma unroll
    for (int nt = 0; nt < 4; nt++)
#pragma unroll
      for (int r = 0; r < 8; r++) {
        float p = __expf(sacc[nt][r] - mrow[r]);
        sacc[nt][r] = p;
        rsum[r] += p;
      }
#pragma unroll
    for (int o = 1; o < 16; o <<= 1)
#pragma unroll
      for (int r = 0; r < 8; r++) rsum[r] += __shfl_xor(rsum[r], o, 32);
#pragma unroll
    for (int r = 0; r < 8; r++) lrow[r] = lrow[r] * fac[r] + rsum[r];
#pragma unroll
    for (int dt = 0; dt < 4; dt++)
#pragma unroll
      for (int r = 0; r < 8; r++) Oacc[dt][r] *= fac[r];

    // Round-trip P through per-wave LDS (C layout -> A-fragment layout)
    {
      unsigned short* pp = (unsigned short*)&Ps[wave * 16 * 33];
#pragma unroll
      for (int nt = 0; nt < 4; nt++)
#pragma unroll
        for (int r = 0; r < 8; r++)
          pp[(r + 8 * lh) * 66 + nt * 16 + li] = f2bf(sacc[nt][r]);
    }

    // O += P @ V : K-dim = 64 keys (2 WMMA k-steps), N = 64 d
#pragma unroll
    for (int ks = 0; ks < 2; ks++) {
      Frag a;
      const unsigned int* pa = &Ps[wave * 16 * 33 + li * 33 + ks * 16];
#pragma unroll
      for (int j = 0; j < 8; j++) a.u[j] = pa[APAIR(j, lh)];
#pragma unroll
      for (int dt = 0; dt < 4; dt++) {
        Frag b;
        const unsigned int* pb = &Vs[(dt * 16 + li) * 33 + ks * 16];
#pragma unroll
        for (int j = 0; j < 8; j++) b.u[j] = pb[BPAIR(j, lh)];
        Oacc[dt] = wmma_bf16(a.v, b.v, Oacc[dt]);
      }
    }
  }

  float inv[8];
#pragma unroll
  for (int r = 0; r < 8; r++) inv[r] = (lrow[r] > 0.f) ? (1.f / lrow[r]) : 0.f;
#pragma unroll
  for (int dt = 0; dt < 4; dt++)
#pragma unroll
    for (int r = 0; r < 8; r++) {
      int qg = q0 + wave * 16 + r + 8 * lh;
      Og[(size_t)qg * D_DIM + h * 64 + dt * 16 + li] = f2bf(Oacc[dt][r] * inv[r]);
    }
}

// ---------------------------------------------------------------------------
// Pooler: out = tanh(x[0] @ pool_w + pool_b) @ fc_w + fc_b  -> [10] f32
// ---------------------------------------------------------------------------
__global__ __launch_bounds__(256) void pool_kernel(
    const float* __restrict__ xf, const float* __restrict__ pw,
    const float* __restrict__ pb, const float* __restrict__ fw,
    const float* __restrict__ fb, float* __restrict__ out) {
  __shared__ float x0[D_DIM];
  __shared__ float pooled[D_DIM];
  for (int i = threadIdx.x; i < D_DIM; i += 256) x0[i] = xf[i];
  __syncthreads();
  for (int i = threadIdx.x; i < D_DIM; i += 256) {
    float s = pb[i];
    for (int k = 0; k < D_DIM; k++) s += x0[k] * pw[(size_t)k * D_DIM + i];
    pooled[i] = tanhf(s);
  }
  __syncthreads();
  if (threadIdx.x < C_CLS) {
    float s = fb[threadIdx.x];
    for (int k = 0; k < D_DIM; k++) s += pooled[k] * fw[(size_t)k * C_CLS + threadIdx.x];
    out[threadIdx.x] = s;
  }
}

// ---------------------------------------------------------------------------
extern "C" void kernel_launch(void* const* d_in, const int* in_sizes, int n_in,
                              void* d_out, int out_size, void* d_ws, size_t ws_size,
                              hipStream_t stream) {
  (void)in_sizes; (void)n_in; (void)out_size; (void)ws_size;
  const int*   ctx  = (const int*)d_in[0];
  const int*   mask = (const int*)d_in[1];
  const float* wemb = (const float*)d_in[2];
  const float* pemb = (const float*)d_in[3];
  const float* elns = (const float*)d_in[4];
  const float* elnb = (const float*)d_in[5];
  const float* Wq   = (const float*)d_in[6];
  const float* bq   = (const float*)d_in[7];
  const float* Wk   = (const float*)d_in[8];
  const float* bk   = (const float*)d_in[9];
  const float* Wv   = (const float*)d_in[10];
  const float* bv   = (const float*)d_in[11];
  const float* Wo   = (const float*)d_in[12];
  const float* bo   = (const float*)d_in[13];
  const float* l1s  = (const float*)d_in[14];
  const float* l1b  = (const float*)d_in[15];
  const float* W1   = (const float*)d_in[16];
  const float* b1   = (const float*)d_in[17];
  const float* W2   = (const float*)d_in[18];
  const float* b2   = (const float*)d_in[19];
  const float* l2s  = (const float*)d_in[20];
  const float* l2b  = (const float*)d_in[21];
  const float* pw   = (const float*)d_in[22];
  const float* pbv  = (const float*)d_in[23];
  const float* fw   = (const float*)d_in[24];
  const float* fb   = (const float*)d_in[25];
  float* out = (float*)d_out;

  char* ws = (char*)d_ws;
  size_t off = 0;
  auto alloc = [&](size_t bytes) -> void* {
    void* p = ws + off;
    off = (off + bytes + 255) & ~(size_t)255;
    return p;
  };
  float*          xf = (float*)alloc((size_t)S_LEN * D_DIM * 4);
  unsigned short* xb = (unsigned short*)alloc((size_t)S_LEN * D_DIM * 2);
  unsigned short* qb = (unsigned short*)alloc((size_t)S_LEN * D_DIM * 2);
  unsigned short* kb = (unsigned short*)alloc((size_t)S_LEN * D_DIM * 2);
  unsigned short* vb = (unsigned short*)alloc((size_t)S_LEN * D_DIM * 2);
  unsigned short* ab = (unsigned short*)alloc((size_t)S_LEN * D_DIM * 2);
  float*          yf = (float*)alloc((size_t)S_LEN * D_DIM * 4);
  unsigned short* hb = (unsigned short*)alloc((size_t)S_LEN * F_DIM * 2);
  unsigned short* wt = (unsigned short*)alloc((size_t)D_DIM * F_DIM * 2);  // weights

  embed_ln_kernel<<<S_LEN, 256, 0, stream>>>(ctx, wemb, pemb, elns, elnb, xf, xb);

  const dim3 gD(D_DIM / 128, S_LEN / 128);        // (6, 32)
  const dim3 gF(F_DIM / 128, S_LEN / 128);        // (24, 32)
  const dim3 gA(S_LEN / 128, 12);                 // (32, 12)
  const dim3 tDD(D_DIM / 32, D_DIM / 32);         // (24, 24)
  const dim3 tDF(D_DIM / 32, F_DIM / 32);         // (24, 96)  K=768,N=3072
  const dim3 tFD(F_DIM / 32, D_DIM / 32);         // (96, 24)  K=3072,N=768

  for (int l = 0; l < 12; l++) {
    const float* wq_ = Wq + (size_t)l * D_DIM * D_DIM;
    const float* wk_ = Wk + (size_t)l * D_DIM * D_DIM;
    const float* wv_ = Wv + (size_t)l * D_DIM * D_DIM;
    const float* wo_ = Wo + (size_t)l * D_DIM * D_DIM;
    const float* w1_ = W1 + (size_t)l * D_DIM * F_DIM;
    const float* w2_ = W2 + (size_t)l * F_DIM * D_DIM;

    wconv_kernel<<<tDD, 256, 0, stream>>>(wq_, wt, D_DIM, D_DIM);
    gemm_kernel<0, true, false><<<gD, 256, 0, stream>>>(
        xb, wt, bq + (size_t)l * D_DIM, nullptr, qb, S_LEN, D_DIM, D_DIM);
    wconv_kernel<<<tDD, 256, 0, stream>>>(wk_, wt, D_DIM, D_DIM);
    gemm_kernel<0, true, false><<<gD, 256, 0, stream>>>(
        xb, wt, bk + (size_t)l * D_DIM, nullptr, kb, S_LEN, D_DIM, D_DIM);
    wconv_kernel<<<tDD, 256, 0, stream>>>(wv_, wt, D_DIM, D_DIM);
    gemm_kernel<0, true, false><<<gD, 256, 0, stream>>>(
        xb, wt, bv + (size_t)l * D_DIM, nullptr, vb, S_LEN, D_DIM, D_DIM);

    attn_kernel<<<gA, 256, 0, stream>>>(qb, kb, vb, mask, ab);

    wconv_kernel<<<tDD, 256, 0, stream>>>(wo_, wt, D_DIM, D_DIM);
    gemm_kernel<0, false, true><<<gD, 256, 0, stream>>>(
        ab, wt, bo + (size_t)l * D_DIM, yf, nullptr, S_LEN, D_DIM, D_DIM);
    add_ln_kernel<<<S_LEN, 256, 0, stream>>>(
        xf, yf, l1s + (size_t)l * D_DIM, l1b + (size_t)l * D_DIM, xb);

    wconv_kernel<<<tDF, 256, 0, stream>>>(w1_, wt, D_DIM, F_DIM);
    gemm_kernel<1, true, false><<<gF, 256, 0, stream>>>(
        xb, wt, b1 + (size_t)l * F_DIM, nullptr, hb, S_LEN, F_DIM, D_DIM);
    wconv_kernel<<<tFD, 256, 0, stream>>>(w2_, wt, F_DIM, D_DIM);
    gemm_kernel<0, false, true><<<gD, 256, 0, stream>>>(
        hb, wt, b2 + (size_t)l * D_DIM, yf, nullptr, S_LEN, D_DIM, F_DIM);
    add_ln_kernel<<<S_LEN, 256, 0, stream>>>(
        xf, yf, l2s + (size_t)l * D_DIM, l2b + (size_t)l * D_DIM, xb);
  }

  pool_kernel<<<1, 256, 0, stream>>>(xf, pw, pbv, fw, fb, out);
}
